// Mamba2LMHeadModel_54185307407024
// MI455X (gfx1250) — matrix operand order repaired and checked by reading, not compiled, not run on hardware
//
#include <hip/hip_runtime.h>
#include <hip/hip_bf16.h>

// ---------------------------------------------------------------------------
// Mamba2 block forward for MI455X (gfx1250, wave32, WMMA bf16 16x16x32).
// ---------------------------------------------------------------------------

typedef __attribute__((ext_vector_type(16))) __bf16 v16bf;
typedef __attribute__((ext_vector_type(8)))  __bf16 v8bf;
typedef __attribute__((ext_vector_type(4)))  __bf16 v4bf;
typedef __attribute__((ext_vector_type(8)))  float  v8f;
typedef __attribute__((ext_vector_type(4)))  float  v4f;

static constexpr int BATCH  = 2;
static constexpr int SEQ    = 4096;
static constexpr int DMODEL = 2048;
static constexpr int DIN    = 4096;            // d_inner
static constexpr int HEADS  = 64;
static constexpr int HD     = 64;              // headdim
static constexpr int DST    = 128;             // d_state
static constexpr int CH     = 64;              // chunk size
static constexpr int NCH    = SEQ / CH;        // 64 chunks
static constexpr int CONVD  = DIN + 2 * DST;   // 4352
static constexpr int DPROJ  = 2 * DIN + 2 * DST + HEADS; // 8512
static constexpr int XBC_OFF = DIN;            // 4096
static constexpr int DT_OFF  = 2 * DIN + 2 * DST; // 8448

#define DEV __device__ __forceinline__

// ---- WMMA bf16 helpers -----------------------------------------------------
// A operand (16x32, M x K), LDS layout [M][K] row-major:
//   lanes 0-15: M=lane, elements 0..7 -> K=k0+0..7,  8..15 -> K=k0+16..23
//   lanes16-31: M=lane-16, elements 0..7 -> K=k0+8..15, 8..15 -> K=k0+24..31
DEV v16bf frag_a(const __bf16* row, int k0, int khalf) {
    const __bf16* p0 = row + k0 + khalf * 8;
    const __bf16* p1 = p0 + 16;
    v16bf v;
#pragma unroll
    for (int e = 0; e < 8; ++e) { v[e] = p0[e]; v[e + 8] = p1[e]; }
    return v;
}
DEV v16bf frag_a_scaled(const __bf16* row, int k0, int khalf, float s) {
    const __bf16* p0 = row + k0 + khalf * 8;
    const __bf16* p1 = p0 + 16;
    v16bf v;
#pragma unroll
    for (int e = 0; e < 8; ++e) {
        v[e]     = (__bf16)((float)p0[e] * s);
        v[e + 8] = (__bf16)((float)p1[e] * s);
    }
    return v;
}
// B operand (32x16, K x N), LDS layout TRANSPOSED [N][K]:
//   lanes 0-15: N=lane, elements 0..7 -> K=k0+0..7,  8..15 -> K=k0+16..23
//   lanes16-31: N=lane-16, elements 0..7 -> K=k0+8..15, 8..15 -> K=k0+24..31
DEV v16bf frag_b(const __bf16* col, int k0, int khalf) {
    const __bf16* p = col + k0 + khalf * 8;
    v16bf v;
#pragma unroll
    for (int e = 0; e < 8; ++e) { v[e] = p[e]; v[e + 8] = p[e + 16]; }
    return v;
}
DEV v8f wmma_bf16(v16bf a, v16bf b, v8f c) {
    return __builtin_amdgcn_wmma_f32_16x16x32_bf16(
        /*neg_a=*/false, a, /*neg_b=*/false, b,
        /*c_mod=*/(short)0, c, /*reuse_a=*/false, /*reuse_b=*/false);
}

DEV float siluf(float x) { return x / (1.f + __expf(-x)); }
DEV v4bf pack4(v4f v) {
    v4bf o;
#pragma unroll
    for (int e = 0; e < 4; ++e) o[e] = (__bf16)v[e];
    return o;
}
DEV v4f ld4(const float* p) { return *(const v4f*)p; }

// ---------------------------------------------------------------------------
// K1: dt softplus, causal depthwise conv(k=4) + SiLU; split x / B / C.
// Vectorized: float4 over channels (region boundaries 0/4096/4224 are 4-aligned)
// ---------------------------------------------------------------------------
__global__ __launch_bounds__(256) void prep_kernel(
    const float* __restrict__ zx, const float* __restrict__ convw,
    const float* __restrict__ convb, const float* __restrict__ dtb,
    const float* __restrict__ Avec,
    float* __restrict__ xws, float* __restrict__ Bws, float* __restrict__ Cws,
    float* __restrict__ dtws, float* __restrict__ adtws)
{
    const int bl  = blockIdx.x;          // b*SEQ + l
    const int l   = bl % SEQ;
    const int tid = threadIdx.x;
    const long rowbase = (long)bl * DPROJ;

    if (tid < HEADS) {
        float v  = zx[rowbase + DT_OFF + tid] + dtb[tid];
        float dt = (v > 20.f) ? v : log1pf(__expf(v));   // softplus
        dtws[(long)bl * HEADS + tid]  = dt;
        adtws[(long)bl * HEADS + tid] = Avec[tid] * dt;
    }
    for (int i = tid; i < CONVD / 4; i += 256) {
        const int ch = i * 4;
        v4f acc = ld4(&convb[ch]);
        v4f w0 = ld4(&convw[(ch + 0) * 4]);   // 4 taps of channel ch+0
        v4f w1 = ld4(&convw[(ch + 1) * 4]);
        v4f w2 = ld4(&convw[(ch + 2) * 4]);
        v4f w3 = ld4(&convw[(ch + 3) * 4]);
#pragma unroll
        for (int j = 0; j < 4; ++j) {
            int ll = l - 3 + j;
            if (ll >= 0) {
                v4f xv = ld4(&zx[rowbase + (long)(ll - l) * DPROJ + XBC_OFF + ch]);
                acc[0] += w0[j] * xv[0];
                acc[1] += w1[j] * xv[1];
                acc[2] += w2[j] * xv[2];
                acc[3] += w3[j] * xv[3];
            }
        }
        v4f s;
#pragma unroll
        for (int e = 0; e < 4; ++e) s[e] = siluf(acc[e]);
        if (ch < DIN)             *(v4f*)&xws[(long)bl * DIN + ch] = s;
        else if (ch < DIN + DST)  *(v4f*)&Bws[(long)bl * DST + (ch - DIN)] = s;
        else                      *(v4f*)&Cws[(long)bl * DST + (ch - DIN - DST)] = s;
    }
}

// ---------------------------------------------------------------------------
// K2: per-chunk inclusive cumsum of A*dt, and per-chunk last value
// ---------------------------------------------------------------------------
__global__ void cumsum_kernel(const float* __restrict__ adt,
                              float* __restrict__ acs, float* __restrict__ alast)
{
    int t = blockIdx.x * blockDim.x + threadIdx.x;   // (b*NCH+c)*HEADS+h
    if (t >= BATCH * NCH * HEADS) return;
    int h = t % HEADS;
    int c = (t / HEADS) % NCH;
    int b = t / (HEADS * NCH);
    float s = 0.f;
    long obase = (long)t * CH;
    for (int l = 0; l < CH; ++l) {
        s += adt[((long)(b * SEQ + c * CH + l)) * HEADS + h];
        acs[obase + l] = s;
    }
    alast[t] = s;
}

// ---------------------------------------------------------------------------
// K3: chunk states  S[n,p] = sum_l (B[l,n]*exp(acs_last-acs[l])) * (x[l,p]*dt[l])
//     128x64 output, K=64, via v_wmma_f32_16x16x32_bf16
// ---------------------------------------------------------------------------
__global__ __launch_bounds__(256) void chunk_states_kernel(
    const float* __restrict__ x, const float* __restrict__ dt,
    const float* __restrict__ Bm, const float* __restrict__ acs,
    float* __restrict__ states)
{
    __shared__ __bf16 Bd[DST][CH];   // A-operand layout [n][l]
    __shared__ __bf16 Xp[HD][CH];    // B-operand layout [p][l]
    __shared__ float  acs_s[CH];

    const int bid = blockIdx.x;                 // (b*NCH+c)*HEADS+h
    const int h = bid % HEADS;
    const int c = (bid / HEADS) % NCH;
    const int b = bid / (HEADS * NCH);
    const int tid = threadIdx.x;
    const long bch = (long)bid;

    if (tid < CH) acs_s[tid] = acs[bch * CH + tid];
    __syncthreads();
    const float last = acs_s[CH - 1];
    const long lrow0 = (long)(b * SEQ + c * CH);

    for (int i = tid; i < CH * DST / 4; i += 256) {
        int l = i >> 5, n4 = (i & 31) * 4;
        v4f bv = ld4(&Bm[(lrow0 + l) * DST + n4]) * __expf(last - acs_s[l]);
#pragma unroll
        for (int e = 0; e < 4; ++e) Bd[n4 + e][l] = (__bf16)bv[e];
    }
    for (int i = tid; i < CH * HD / 4; i += 256) {
        int l = i >> 4, p4 = (i & 15) * 4;
        float d = dt[(lrow0 + l) * HEADS + h];
        v4f xv = ld4(&x[((lrow0 + l) * HEADS + h) * HD + p4]) * d;
#pragma unroll
        for (int e = 0; e < 4; ++e) Xp[p4 + e][l] = (__bf16)xv[e];
    }
    __syncthreads();

    const int wave = tid >> 5, lane = tid & 31;
    const int mrow = lane & 15, khalf = lane >> 4;
    const int rowoff = (lane >= 16) ? 8 : 0;
    v8f acc[4] = {};
#pragma unroll
    for (int k0 = 0; k0 < CH; k0 += 32) {
        v16bf a = frag_a(&Bd[wave * 16 + mrow][0], k0, khalf);
#pragma unroll
        for (int tj = 0; tj < 4; ++tj) {
            v16bf bb = frag_b(&Xp[tj * 16 + mrow][0], k0, khalf);
            acc[tj] = wmma_bf16(a, bb, acc[tj]);
        }
    }
    const long sbase = bch * (DST * HD);
#pragma unroll
    for (int tj = 0; tj < 4; ++tj)
#pragma unroll
        for (int r = 0; r < 8; ++r) {
            int n = wave * 16 + r + rowoff;
            int p = tj * 16 + mrow;
            states[sbase + n * HD + p] = acc[tj][r];
        }
}

// ---------------------------------------------------------------------------
// K4: inter-chunk scan (in place: states[c] becomes the "incoming" state),
//     writes final_state (b,h,p,n) to output tail. float4 over p.
// ---------------------------------------------------------------------------
__global__ void scan_kernel(float* __restrict__ states,
                            const float* __restrict__ alast,
                            float* __restrict__ finalst)
{
    int t = blockIdx.x * blockDim.x + threadIdx.x;
    if (t >= BATCH * HEADS * HD * DST / 4) return;
    int p4 = (t & 15) * 4;
    int n  = (t >> 4) & 127;
    int h  = (t >> 11) & 63;
    int b  = t >> 17;
    v4f S = {0.f, 0.f, 0.f, 0.f};
    for (int c = 0; c < NCH; ++c) {
        long bch = ((long)(b * NCH + c)) * HEADS + h;
        long idx = bch * (DST * HD) + n * HD + p4;
        v4f cur = *(v4f*)&states[idx];
        *(v4f*)&states[idx] = S;               // state entering this chunk
        S = S * __expf(alast[bch]) + cur;
    }
#pragma unroll
    for (int e = 0; e < 4; ++e)
        finalst[(((long)(b * HEADS + h)) * HD + (p4 + e)) * DST + n] = S[e];
}

// ---------------------------------------------------------------------------
// K5: Y = mask(C·Bᵀ)·Xdt + (C·e^{acs})·S_in ; += D*x ; *= silu(z). In-place x->y.
// ---------------------------------------------------------------------------
__global__ __launch_bounds__(256) void chunk_y_kernel(
    const float* __restrict__ dt, const float* __restrict__ Bm,
    const float* __restrict__ Cm, const float* __restrict__ acs,
    const float* __restrict__ states, const float* __restrict__ zx,
    const float* __restrict__ Dvec, float* __restrict__ xy)
{
    __shared__ __bf16 Cn[CH][DST];   // A-operand [l][n]
    __shared__ __bf16 Bt[CH][DST];   // B-operand [s][n]; reused as M[l][s] later
    __shared__ __bf16 Xp[HD][CH];    // B-operand [p][s]
    __shared__ __bf16 Sp[HD][DST];   // B-operand [p][n]
    __shared__ float  acs_s[CH];

    const int bid = blockIdx.x;
    const int h = bid % HEADS;
    const int c = (bid / HEADS) % NCH;
    const int b = bid / (HEADS * NCH);
    const int tid = threadIdx.x;
    const long bch = (long)bid;

    if (tid < CH) acs_s[tid] = acs[bch * CH + tid];
    __syncthreads();
    const long lrow0 = (long)(b * SEQ + c * CH);

    for (int i = tid; i < CH * DST / 4; i += 256) {
        int l = i >> 5, n4 = (i & 31) * 4;
        *(v4bf*)&Cn[l][n4] = pack4(ld4(&Cm[(lrow0 + l) * DST + n4]));
        *(v4bf*)&Bt[l][n4] = pack4(ld4(&Bm[(lrow0 + l) * DST + n4]));
    }
    for (int i = tid; i < CH * HD / 4; i += 256) {
        int l = i >> 4, p4 = (i & 15) * 4;
        float d = dt[(lrow0 + l) * HEADS + h];
        v4f xv = ld4(&xy[((lrow0 + l) * HEADS + h) * HD + p4]) * d;
#pragma unroll
        for (int e = 0; e < 4; ++e) Xp[p4 + e][l] = (__bf16)xv[e];
    }
    for (int i = tid; i < DST * HD / 4; i += 256) {
        int n = i >> 4, p4 = (i & 15) * 4;
        v4f sv = ld4(&states[bch * (DST * HD) + n * HD + p4]);
#pragma unroll
        for (int e = 0; e < 4; ++e) Sp[p4 + e][n] = (__bf16)sv[e];
    }
    __syncthreads();

    const int wave = tid >> 5, lane = tid & 31;
    const int mrow = lane & 15, khalf = lane >> 4;
    const int rowoff = (lane >= 16) ? 8 : 0;

    // GEMM1: G = C (64x128) x Bᵀ (128x64). Each wave: 2 of 16 tiles.
    v8f g[2] = {};
#pragma unroll
    for (int u = 0; u < 2; ++u) {
        int t = wave * 2 + u, ti = t >> 2, tj = t & 3;
#pragma unroll
        for (int k0 = 0; k0 < DST; k0 += 32) {
            v16bf a  = frag_a(&Cn[ti * 16 + mrow][0], k0, khalf);
            v16bf bb = frag_b(&Bt[tj * 16 + mrow][0], k0, khalf);
            g[u] = wmma_bf16(a, bb, g[u]);
        }
    }
    __syncthreads();                      // all waves done reading Bt
    __bf16* M = &Bt[0][0];                // reuse as M[l][s], 64x64
#pragma unroll
    for (int u = 0; u < 2; ++u) {
        int t = wave * 2 + u, ti = t >> 2, tj = t & 3;
#pragma unroll
        for (int r = 0; r < 8; ++r) {
            int lr = ti * 16 + r + rowoff;
            int s  = tj * 16 + mrow;
            float f = (s <= lr) ? __expf(acs_s[lr] - acs_s[s]) : 0.f;
            M[lr * CH + s] = (__bf16)(g[u][r] * f);
        }
    }
    __syncthreads();

    // GEMM2: Y = M x Xdt (K=64) + (C scaled by e^{acs[l]}) x S_in (K=128)
#pragma unroll
    for (int u = 0; u < 2; ++u) {
        int t = wave * 2 + u, ti = t >> 2, tj = t & 3;
        v8f acc = {};
#pragma unroll
        for (int k0 = 0; k0 < CH; k0 += 32) {
            v16bf a  = frag_a(M + (ti * 16 + mrow) * CH, k0, khalf);
            v16bf bb = frag_b(&Xp[tj * 16 + mrow][0], k0, khalf);
            acc = wmma_bf16(a, bb, acc);
        }
        float rscale = __expf(acs_s[ti * 16 + mrow]);
#pragma unroll
        for (int k0 = 0; k0 < DST; k0 += 32) {
            v16bf a  = frag_a_scaled(&Cn[ti * 16 + mrow][0], k0, khalf, rscale);
            v16bf bb = frag_b(&Sp[tj * 16 + mrow][0], k0, khalf);
            acc = wmma_bf16(a, bb, acc);
        }
#pragma unroll
        for (int r = 0; r < 8; ++r) {
            int lr = ti * 16 + r + rowoff;
            int p  = tj * 16 + mrow;
            long gi = ((lrow0 + lr) * HEADS + h) * HD + p;
            float xv = xy[gi];
            float y  = acc[r] + Dvec[h] * xv;
            float z  = zx[(lrow0 + lr) * DPROJ + h * HD + p];
            y *= siluf(z);
            xy[gi] = y;                    // in-place: x -> gated y
        }
    }
}

// ---------------------------------------------------------------------------
// K6: RMSNorm over d_inner, write bf16 for the out-projection GEMM
// ---------------------------------------------------------------------------
__global__ __launch_bounds__(256) void rmsnorm_kernel(
    const float* __restrict__ y, const float* __restrict__ w,
    __bf16* __restrict__ out)
{
    __shared__ float red[256];
    const int bl = blockIdx.x, tid = threadIdx.x;
    const long base = (long)bl * DIN;
    float s = 0.f;
    for (int i = tid; i < DIN / 4; i += 256) {
        v4f v = ld4(&y[base + i * 4]);
        s += v[0] * v[0] + v[1] * v[1] + v[2] * v[2] + v[3] * v[3];
    }
    red[tid] = s;
    __syncthreads();
    for (int o = 128; o > 0; o >>= 1) {
        if (tid < o) red[tid] += red[tid + o];
        __syncthreads();
    }
    float inv = rsqrtf(red[0] / (float)DIN + 1e-5f);
    for (int i = tid; i < DIN / 4; i += 256) {
        v4f v = ld4(&y[base + i * 4]) * inv;
        v4f ww = ld4(&w[i * 4]);
        *(v4bf*)&out[base + i * 4] = pack4(v * ww);
    }
}

// ---------------------------------------------------------------------------
// K7: out = Y(8192x4096,bf16) x Wᵀ(4096x2048) + bias. 128x128 block tiles.
// ---------------------------------------------------------------------------
__global__ __launch_bounds__(256) void outproj_kernel(
    const __bf16* __restrict__ Y, const float* __restrict__ W,
    const float* __restrict__ bias, float* __restrict__ out)
{
    __shared__ __bf16 As[128][64];   // A-operand [m][k]
    __shared__ __bf16 Bs[128][64];   // B-operand [n][k] (== W row-major slice)
    const int m0 = blockIdx.x * 128;
    const int n0 = blockIdx.y * 128;
    const int tid = threadIdx.x;
    const int wave = tid >> 5, lane = tid & 31;
    const int wr = wave & 3, wc = wave >> 2;       // wave tile: 32 rows x 64 cols
    const int mrow = lane & 15, khalf = lane >> 4;
    const int rowoff = (lane >= 16) ? 8 : 0;
    v8f acc[2][4] = {};

    for (int k0 = 0; k0 < DIN; k0 += 64) {
        __syncthreads();
        for (int i = tid; i < 128 * 8; i += 256) {        // 16B copies of Y panel
            int r = i >> 3, k8 = (i & 7) * 8;
            *(v8bf*)&As[r][k8] = *(const v8bf*)&Y[(long)(m0 + r) * DIN + k0 + k8];
        }
        for (int i = tid; i < 128 * 16; i += 256) {       // float4 -> 4xbf16 of W panel
            int r = i >> 4, k4 = (i & 15) * 4;
            *(v4bf*)&Bs[r][k4] = pack4(ld4(&W[(long)(n0 + r) * DIN + k0 + k4]));
        }
        if (k0 + 64 < DIN) {   // global_prefetch_b8 of next K panels
            __builtin_prefetch(&Y[(long)(m0 + (tid >> 1)) * DIN + k0 + 64 + (tid & 1) * 32], 0, 1);
            __builtin_prefetch(&W[(long)(n0 + (tid >> 1)) * DIN + k0 + 64 + (tid & 1) * 16], 0, 1);
        }
        __syncthreads();
#pragma unroll
        for (int kk = 0; kk < 64; kk += 32) {
            v16bf a0 = frag_a(&As[wr * 32 + mrow][0], kk, khalf);
            v16bf a1 = frag_a(&As[wr * 32 + 16 + mrow][0], kk, khalf);
#pragma unroll
            for (int tj = 0; tj < 4; ++tj) {
                v16bf bb = frag_b(&Bs[wc * 64 + tj * 16 + mrow][0], kk, khalf);
                acc[0][tj] = wmma_bf16(a0, bb, acc[0][tj]);
                acc[1][tj] = wmma_bf16(a1, bb, acc[1][tj]);
            }
        }
    }
#pragma unroll
    for (int ti = 0; ti < 2; ++ti)
#pragma unroll
        for (int tj = 0; tj < 4; ++tj)
#pragma unroll
            for (int r = 0; r < 8; ++r) {
                int row = m0 + wr * 32 + ti * 16 + r + rowoff;
                int col = n0 + wc * 64 + tj * 16 + mrow;
                out[(long)row * DMODEL + col] = acc[ti][tj][r] + bias[col];
            }
}

// ---------------------------------------------------------------------------
extern "C" void kernel_launch(void* const* d_in, const int* in_sizes, int n_in,
                              void* d_out, int out_size, void* d_ws, size_t ws_size,
                              hipStream_t stream)
{
    (void)in_sizes; (void)n_in; (void)out_size; (void)ws_size;
    const float* zx  = (const float*)d_in[0];
    const float* cw  = (const float*)d_in[1];
    const float* cb  = (const float*)d_in[2];
    const float* dtb = (const float*)d_in[3];
    const float* Av  = (const float*)d_in[4];
    const float* Dv  = (const float*)d_in[5];
    const float* rw  = (const float*)d_in[6];
    const float* Wo  = (const float*)d_in[7];
    const float* bo  = (const float*)d_in[8];
    float* out       = (float*)d_out;
    float* final_out = out + (size_t)BATCH * SEQ * DMODEL;

    char* ws = (char*)d_ws;
    size_t off = 0;
    auto alloc = [&](size_t bytes) -> char* {
        char* p = ws + off;
        off += (bytes + 255) & ~(size_t)255;
        return p;
    };
    float*  xws   = (float*) alloc((size_t)BATCH * SEQ * DIN * 4);     // x -> y in place
    float*  Bws   = (float*) alloc((size_t)BATCH * SEQ * DST * 4);
    float*  Cws   = (float*) alloc((size_t)BATCH * SEQ * DST * 4);
    float*  dtws  = (float*) alloc((size_t)BATCH * SEQ * HEADS * 4);
    float*  adtws = (float*) alloc((size_t)BATCH * SEQ * HEADS * 4);
    float*  acsws = (float*) alloc((size_t)BATCH * NCH * HEADS * CH * 4);
    float*  alast = (float*) alloc((size_t)BATCH * NCH * HEADS * 4);
    float*  stws  = (float*) alloc((size_t)BATCH * NCH * HEADS * DST * HD * 4);
    __bf16* ynorm = (__bf16*)alloc((size_t)BATCH * SEQ * DIN * 2);

    prep_kernel<<<BATCH * SEQ, 256, 0, stream>>>(zx, cw, cb, dtb, Av,
                                                 xws, Bws, Cws, dtws, adtws);
    cumsum_kernel<<<(BATCH * NCH * HEADS + 255) / 256, 256, 0, stream>>>(adtws, acsws, alast);
    chunk_states_kernel<<<BATCH * NCH * HEADS, 256, 0, stream>>>(xws, dtws, Bws, acsws, stws);
    scan_kernel<<<(BATCH * HEADS * HD * DST / 4 + 255) / 256, 256, 0, stream>>>(stws, alast, final_out);
    chunk_y_kernel<<<BATCH * NCH * HEADS, 256, 0, stream>>>(dtws, Bws, Cws, acsws,
                                                            stws, zx, Dv, xws);
    rmsnorm_kernel<<<BATCH * SEQ, 256, 0, stream>>>(xws, rw, ynorm);
    dim3 og(BATCH * SEQ / 128, DMODEL / 128);
    outproj_kernel<<<og, 256, 0, stream>>>(ynorm, Wo, bo, out);
}
